// MaterializedMultiHeadAttention_61727269978150
// MI455X (gfx1250) — matrix operand order repaired and checked by reading, not compiled, not run on hardware
//
#include <hip/hip_runtime.h>

// ---- problem constants (B=1) ----
constexpr int kN   = 384;            // nodes
constexpr int kNN  = kN * kN;        // 147456
constexpr int kE   = 64;             // EDGE_IN / EDGE_OUT
constexpr int kHA  = 256;            // H*A = H*V
constexpr int kNodeIn  = 128;
constexpr int kNodeOut = 128;

typedef __attribute__((ext_vector_type(16))) __bf16 v16bf;
typedef __attribute__((ext_vector_type(8)))  float  v8f;

// ---------------- helpers ----------------
__device__ inline unsigned short f2bf(float f) {
  unsigned u = __float_as_uint(f);
  u += 0x7fffu + ((u >> 16) & 1u);      // round-to-nearest-even
  return (unsigned short)(u >> 16);
}

__device__ inline v16bf ld_frag2(const unsigned short* p0, const unsigned short* p1) {
  union { uint4 q[2]; v16bf v; } u;
  u.q[0] = *reinterpret_cast<const uint4*>(p0);
  u.q[1] = *reinterpret_cast<const uint4*>(p1);
  return u.v;
}

// A fragment: 16x32 bf16 tile at (row0,k0) of row-major [.,stride] matrix.
// ISA layout: lanes 0-15 row M=l hold K {k0..k0+7, k0+16..k0+23};
//             lanes 16-31 hold K {k0+8..15, k0+24..31}.
__device__ inline v16bf ld_a(const unsigned short* m, int stride, int row0, int k0,
                             int l, int hi) {
  const unsigned short* p = m + (size_t)(row0 + l) * stride + k0 + hi * 8;
  return ld_frag2(p, p + 16);
}

// B fragment: logical B[k][n] stored transposed as mT[n][k] (row-major, stride).
// ISA layout: lanes 0-15 col N=l hold K {k0..k0+7, k0+16..k0+23}; lanes 16-31 K {k0+8..15, k0+24..31}.
__device__ inline v16bf ld_b(const unsigned short* mT, int stride, int k0, int n0,
                             int l, int hi) {
  const unsigned short* p = mT + (size_t)(n0 + l) * stride + k0 + hi * 8;
  return ld_frag2(p, p + 16);
}

__device__ inline v8f wmma_bf16(v16bf a, v16bf b, v8f c) {
  return __builtin_amdgcn_wmma_f32_16x16x32_bf16(false, a, false, b, (short)0, c,
                                                 false, false);
}

// Synchronous staging of edges[:, i, :] into LDS transposed [j][c] bf16 (48 KB).
// float4 loads contiguous in j -> coalesced; 4 bf16 transposed stores each.
__device__ inline void fill_erowT(unsigned short* sE, const float* __restrict__ edges,
                                  int i) {
  for (int q = threadIdx.x; q < kE * (kN / 4); q += 256) {
    int c  = q / (kN / 4);
    int j4 = (q - c * (kN / 4)) * 4;
    const float* gp = edges + (size_t)c * kNN + (size_t)i * kN + j4;
    const float4 v = *reinterpret_cast<const float4*>(gp);
    sE[(j4 + 0) * kE + c] = f2bf(v.x);
    sE[(j4 + 1) * kE + c] = f2bf(v.y);
    sE[(j4 + 2) * kE + c] = f2bf(v.z);
    sE[(j4 + 3) * kE + c] = f2bf(v.w);
    __builtin_prefetch(gp + kN, 0, 0);   // warm next row i+1 (global_prefetch_b8)
  }
}

// ---------------- prep kernels (tiny, VALU) ----------------
// Qs = (W_q x nodes)/sqrt(A); Knode = W_k[:, :128] x nodes; Vnode = W_v[:, :128] x nodes
__global__ void prep_proj_kernel(const float* __restrict__ nodes,
                                 const float* __restrict__ W_q,
                                 const float* __restrict__ W_k,
                                 const float* __restrict__ W_v,
                                 float* __restrict__ Qs,
                                 float* __restrict__ Kn,
                                 float* __restrict__ Vn) {
  int t = blockIdx.x * 256 + threadIdx.x;
  if (t >= 3 * kHA * kN) return;
  int o = t / kN, n = t % kN;
  int which = o >> 8, oo = o & 255;
  float s = 0.f;
  if (which == 0) {
    for (int d = 0; d < kNodeIn; ++d) s += W_q[oo * kNodeIn + d] * nodes[d * kN + n];
    Qs[oo * kN + n] = s * 0.17677669529663687f;  // 1/sqrt(32)
  } else if (which == 1) {
    for (int d = 0; d < kNodeIn; ++d) s += W_k[oo * 192 + d] * nodes[d * kN + n];
    Kn[oo * kN + n] = s;
  } else {
    for (int d = 0; d < kNodeIn; ++d) s += W_v[oo * 192 + d] * nodes[d * kN + n];
    Vn[oo * kN + n] = s;
  }
}

// snode[h,i] = sum_a Qs[h*32+a,i] * Knode[h*32+a,i]
__global__ void prep_snode_kernel(const float* __restrict__ Qs,
                                  const float* __restrict__ Kn,
                                  float* __restrict__ snode) {
  int t = blockIdx.x * 256 + threadIdx.x;
  if (t >= 8 * kN) return;
  int h = t / kN, i = t % kN;
  float s = 0.f;
  for (int a = 0; a < 32; ++a)
    s += Qs[(h * 32 + a) * kN + i] * Kn[(h * 32 + a) * kN + i];
  snode[t] = s;
}

// bf16 conversions of the edge-channel weight slices
__global__ void conv_w_kernel(const float* __restrict__ W_k,
                              const float* __restrict__ W_v,
                              const float* __restrict__ W_e,
                              unsigned short* __restrict__ WkeB,
                              unsigned short* __restrict__ WveB,
                              unsigned short* __restrict__ WeeB) {
  int t = blockIdx.x * 256 + threadIdx.x;
  if (t < 16384) {
    int o = t >> 6, c = t & 63;
    WkeB[t] = f2bf(W_k[o * 192 + 128 + c]);
  } else if (t < 32768) {
    int t2 = t - 16384; int o = t2 >> 6, c = t2 & 63;
    WveB[t2] = f2bf(W_v[o * 192 + 128 + c]);
  } else if (t < 36864) {
    int t2 = t - 32768; int o = t2 >> 6, c = t2 & 63;
    WeeB[t2] = f2bf(W_e[o * 320 + 256 + c]);
  }
}

// ---------------- fused attention over one row i ----------------
__global__ __launch_bounds__(256) void attn_row_kernel(
    const float* __restrict__ edges, const unsigned char* __restrict__ mask,
    const float* __restrict__ Qs, const float* __restrict__ snode,
    const float* __restrict__ Vnode,
    const unsigned short* __restrict__ WkeB, const unsigned short* __restrict__ WveB,
    float* __restrict__ NF) {
  __shared__ alignas(16) unsigned short sE[kE * kN];  // edges[:,i,:]^T bf16, 48 KB
  __shared__ alignas(16) float sP[8 * kN];            // sim row -> probabilities, 12 KB
  const int i = blockIdx.x;
  fill_erowT(sE, edges, i);
  __syncthreads();

  const int w = threadIdx.x >> 5;      // wave == head
  const int lane = threadIdx.x & 31;
  const int l = lane & 15;
  const int hi = lane >> 4;

  // ---- phase 1: sim[w][j] = snode + sum_a q * keyE ----
  v16bf ak[2][2];
#pragma unroll
  for (int m2 = 0; m2 < 2; ++m2)
#pragma unroll
    for (int ks = 0; ks < 2; ++ks)
      ak[m2][ks] = ld_a(WkeB, kE, (2 * w + m2) * 16, ks * 32, l, hi);

  float q0[8], q1[8];
#pragma unroll
  for (int r = 0; r < 8; ++r) {
    q0[r] = Qs[(32 * w + hi * 8 + r) * kN + i];
    q1[r] = Qs[(32 * w + 16 + hi * 8 + r) * kN + i];
  }

  for (int nt = 0; nt < 24; ++nt) {
    int n0 = nt * 16;
    v16bf b0 = ld_b(sE, kE, 0, n0, l, hi);
    v16bf b1 = ld_b(sE, kE, 32, n0, l, hi);
    v8f d0 = {}; v8f d1 = {};
    d0 = wmma_bf16(ak[0][0], b0, d0);
    d0 = wmma_bf16(ak[0][1], b1, d0);
    d1 = wmma_bf16(ak[1][0], b0, d1);
    d1 = wmma_bf16(ak[1][1], b1, d1);
    float s = 0.f;
#pragma unroll
    for (int r = 0; r < 8; ++r) s += q0[r] * d0[r] + q1[r] * d1[r];
    s += __shfl_xor(s, 16, 32);          // combine M 0..7 with M 8..15 halves
    if (hi == 0) sP[w * kN + n0 + l] = s;
  }

  // ---- masked softmax per head (wave-local) ----
  const float sn = snode[w * kN + i];
  const bool maski = mask[i] != 0;
  float mx = -__builtin_inff();
  for (int j = lane; j < kN; j += 32) {
    float s = sP[w * kN + j] + sn;
    sP[w * kN + j] = s;
    if (mask[j]) mx = fmaxf(mx, s);
  }
#pragma unroll
  for (int o = 16; o >= 1; o >>= 1) mx = fmaxf(mx, __shfl_xor(mx, o, 32));
  const bool any = (mx > -__builtin_inff());
  float Z = 0.f;
  for (int j = lane; j < kN; j += 32) {
    float e = (any && mask[j]) ? __expf(sP[w * kN + j] - mx) : 0.f;
    sP[w * kN + j] = e;
    Z += e;
  }
#pragma unroll
  for (int o = 16; o >= 1; o >>= 1) Z += __shfl_xor(Z, o, 32);
  const float inv = (maski && Z > 0.f) ? 1.f / Z : 0.f;
  for (int j = lane; j < kN; j += 32) sP[w * kN + j] *= inv;

  // ---- phase 2: nf[o][i] = sum_j p[w][j] * (valueE + Vnode) ----
#pragma unroll
  for (int m2 = 0; m2 < 2; ++m2)
#pragma unroll
    for (int ks = 0; ks < 2; ++ks)
      ak[m2][ks] = ld_a(WveB, kE, (2 * w + m2) * 16, ks * 32, l, hi);

  float vn0[8], vn1[8];
#pragma unroll
  for (int r = 0; r < 8; ++r) {
    vn0[r] = Vnode[(32 * w + hi * 8 + r) * kN + i];
    vn1[r] = Vnode[(32 * w + 16 + hi * 8 + r) * kN + i];
  }
  float acc0[8] = {}, acc1[8] = {};
  for (int nt = 0; nt < 24; ++nt) {
    int n0 = nt * 16;
    v16bf b0 = ld_b(sE, kE, 0, n0, l, hi);
    v16bf b1 = ld_b(sE, kE, 32, n0, l, hi);
    v8f d0 = {}; v8f d1 = {};
    d0 = wmma_bf16(ak[0][0], b0, d0);
    d0 = wmma_bf16(ak[0][1], b1, d0);
    d1 = wmma_bf16(ak[1][0], b0, d1);
    d1 = wmma_bf16(ak[1][1], b1, d1);
    float pl = sP[w * kN + n0 + l];
#pragma unroll
    for (int r = 0; r < 8; ++r) {
      acc0[r] += pl * (d0[r] + vn0[r]);
      acc1[r] += pl * (d1[r] + vn1[r]);
    }
  }
  // reduce over the 16 lanes of each half (each lane held distinct j columns)
#pragma unroll
  for (int o = 8; o >= 1; o >>= 1) {
#pragma unroll
    for (int r = 0; r < 8; ++r) {
      acc0[r] += __shfl_xor(acc0[r], o, 32);
      acc1[r] += __shfl_xor(acc1[r], o, 32);
    }
  }
  if (l == 0) {
#pragma unroll
    for (int r = 0; r < 8; ++r) {
      NF[(32 * w + hi * 8 + r) * kN + i] = acc0[r];
      NF[(32 * w + 16 + hi * 8 + r) * kN + i] = acc1[r];
    }
  }
}

// ---------------- small output GEMMs (VALU, ~38 MFLOP total) ----------------
// node_out = W_o x NF ; G = 0.5 * W_e[:, :256] x NF
__global__ void small_gemms_kernel(const float* __restrict__ W_o,
                                   const float* __restrict__ W_e,
                                   const float* __restrict__ NF,
                                   float* __restrict__ node_out,
                                   float* __restrict__ G) {
  int t = blockIdx.x * 256 + threadIdx.x;
  if (t < kNodeOut * kN) {
    int o = t / kN, n = t % kN;
    float s = 0.f;
    for (int d = 0; d < kHA; ++d) s += W_o[o * kHA + d] * NF[d * kN + n];
    node_out[t] = s;
  } else if (t < (kNodeOut + kE) * kN) {
    int t2 = t - kNodeOut * kN;
    int o = t2 / kN, n = t2 % kN;
    float s = 0.f;
    for (int d = 0; d < kHA; ++d) s += W_e[o * 320 + d] * NF[d * kN + n];
    G[t2] = 0.5f * s;
  }
}

// ---------------- edge_out row kernel (WMMA + async LDS staging) ----------------
// edge_out[o,i,j] = G[o,i] + G[o,j] + (W_e[:,256:] x edges[:,i,j])[o]
constexpr int kJC = 24;                  // j-columns per async chunk (384/24 = 16 chunks)
constexpr int kChunkElems = kE * kJC;    // 1536 dwords -> 6 async loads per thread

__global__ __launch_bounds__(256) void edge_row_kernel(
    const float* __restrict__ edges, const unsigned short* __restrict__ WeeB,
    const float* __restrict__ G, float* __restrict__ edge_out) {
  __shared__ alignas(16) unsigned short sE[kE * kN];      // 48 KB bf16 [j][c]
  __shared__ alignas(16) float sStage[2][kChunkElems];    // 2 x 6 KB fp32 staging
  const int i = blockIdx.x;

  // --- double-buffered async DMA pipeline: global fp32 -> LDS stage -> bf16 sE ---
  auto issue = [&](int ch, int b) {
#pragma unroll
    for (int r = 0; r < 6; ++r) {
      int e = threadIdx.x + r * 256;
      int c = e / kJC, jj = e - c * kJC;
      const float* g = edges + (size_t)c * kNN + (size_t)i * kN + ch * kJC + jj;
      unsigned ldsoff = (unsigned)(size_t)&sStage[b][e];   // low 32 bits = LDS offset
      asm volatile("global_load_async_to_lds_b32 %0, %1, off"
                   :: "v"(ldsoff), "v"(g) : "memory");
    }
  };
  issue(0, 0);
  int buf = 0;
  for (int ch = 0; ch < kN / kJC; ++ch) {
    if (ch + 1 < kN / kJC) {
      issue(ch + 1, buf ^ 1);
      // async loads complete in order: <=6 outstanding == previous chunk done
      asm volatile("s_wait_asynccnt 0x6" ::: "memory");
    } else {
      asm volatile("s_wait_asynccnt 0x0" ::: "memory");
    }
    __syncthreads();
#pragma unroll
    for (int r = 0; r < 6; ++r) {
      int e = threadIdx.x + r * 256;
      int c = e / kJC, jj = e - c * kJC;
      sE[(ch * kJC + jj) * kE + c] = f2bf(sStage[buf][e]);
    }
    __syncthreads();
    buf ^= 1;
  }

  const int w = threadIdx.x >> 5;
  const int lane = threadIdx.x & 31;
  const int l = lane & 15, hi = lane >> 4;
  const int mt = w & 3;                 // output o-tile
  const int ntBase = (w >> 2) * 12;     // waves 0-3: j-tiles 0-11, waves 4-7: 12-23

  v16bf a0 = ld_a(WeeB, kE, mt * 16, 0, l, hi);
  v16bf a1 = ld_a(WeeB, kE, mt * 16, 32, l, hi);
  float gi[8];
#pragma unroll
  for (int r = 0; r < 8; ++r) gi[r] = G[(mt * 16 + hi * 8 + r) * kN + i];

  for (int nt = ntBase; nt < ntBase + 12; ++nt) {
    int n0 = nt * 16;
    int j = n0 + l;
    v16bf b0 = ld_b(sE, kE, 0, n0, l, hi);
    v16bf b1 = ld_b(sE, kE, 32, n0, l, hi);
    v8f d = {};
    d = wmma_bf16(a0, b0, d);
    d = wmma_bf16(a1, b1, d);
#pragma unroll
    for (int r = 0; r < 8; ++r) {
      int o = mt * 16 + hi * 8 + r;
      edge_out[(size_t)o * kNN + (size_t)i * kN + j] = d[r] + gi[r] + G[o * kN + j];
    }
  }
}

// ---------------- launch ----------------
extern "C" void kernel_launch(void* const* d_in, const int* in_sizes, int n_in,
                              void* d_out, int out_size, void* d_ws, size_t ws_size,
                              hipStream_t stream) {
  (void)in_sizes; (void)n_in; (void)out_size; (void)ws_size;
  const float* nodes = (const float*)d_in[0];            // [1,128,384]
  const float* edges = (const float*)d_in[1];            // [1,64,384,384]
  const unsigned char* mask = (const unsigned char*)d_in[2];  // [1,384] bool
  const float* W_q = (const float*)d_in[3];              // [256,128]
  const float* W_k = (const float*)d_in[4];              // [256,192]
  const float* W_v = (const float*)d_in[5];              // [256,192]
  const float* W_o = (const float*)d_in[6];              // [128,256]
  const float* W_e = (const float*)d_in[7];              // [64,320]

  float* node_out = (float*)d_out;                       // [128,384]
  float* edge_out = node_out + kNodeOut * kN;            // [64,384,384]

  // workspace carve-up (~1.8 MB)
  float* wsf   = (float*)d_ws;
  float* Qs    = wsf;                    // 256*384
  float* Kn    = Qs + kHA * kN;          // 256*384
  float* Vn    = Kn + kHA * kN;          // 256*384
  float* NF    = Vn + kHA * kN;          // 256*384
  float* G     = NF + kHA * kN;          // 64*384
  float* snode = G + kE * kN;            // 8*384
  unsigned short* WkeB = (unsigned short*)(snode + 8 * kN);  // 256*64 (16B aligned)
  unsigned short* WveB = WkeB + kHA * kE;                    // 256*64
  unsigned short* WeeB = WveB + kHA * kE;                    // 64*64

  prep_proj_kernel<<<(3 * kHA * kN) / 256, 256, 0, stream>>>(nodes, W_q, W_k, W_v,
                                                             Qs, Kn, Vn);
  conv_w_kernel<<<36864 / 256, 256, 0, stream>>>(W_k, W_v, W_e, WkeB, WveB, WeeB);
  prep_snode_kernel<<<(8 * kN) / 256, 256, 0, stream>>>(Qs, Kn, snode);
  attn_row_kernel<<<kN, 256, 0, stream>>>(edges, mask, Qs, snode, Vn, WkeB, WveB, NF);
  small_gemms_kernel<<<((kNodeOut + kE) * kN) / 256, 256, 0, stream>>>(W_o, W_e, NF,
                                                                       node_out, G);
  edge_row_kernel<<<kN, 256, 0, stream>>>(edges, WeeB, G, edge_out);
}
